// SAGE_58222576664681
// MI455X (gfx1250) — compile-verified, matrix operand order checked by reading, not run on hardware
//
#include <hip/hip_runtime.h>
#include <hip/hip_bf16.h>

#define N_NODES  50000
#define N_EDGES  800000
#define HIDDEN   128
#define N_LAYERS 4
#define N_GRAPHS 1024
#define BN_EPS   1e-5f
#define ROW_TILES ((N_NODES + 15) / 16)   // 3125

typedef __attribute__((ext_vector_type(16))) __bf16 v16bf;
typedef __attribute__((ext_vector_type(8)))  float  v8f;

// ---------------------------------------------------------------------------
// AtomEncoder: h[n,c] = sum_f emb[x[n,f] + off[f], c]
// ---------------------------------------------------------------------------
__global__ __launch_bounds__(128) void atom_encode_kernel(
    const int* __restrict__ x, const float* __restrict__ emb,
    float* __restrict__ h) {
  const int offs[9] = {0, 119, 123, 135, 147, 157, 163, 169, 171};
  int n = blockIdx.x;
  int c = threadIdx.x;
  float s = 0.0f;
#pragma unroll
  for (int f = 0; f < 9; ++f) {
    int idx = x[n * 9 + f] + offs[f];
    s += emb[(size_t)idx * HIDDEN + c];
  }
  h[(size_t)n * HIDDEN + c] = s;
}

// ---------------------------------------------------------------------------
// Degree count (once; edges are layer-invariant) then invert.
// ---------------------------------------------------------------------------
__global__ __launch_bounds__(256) void deg_count_kernel(
    const int* __restrict__ ei, float* __restrict__ deg) {
  int e = blockIdx.x * blockDim.x + threadIdx.x;
  if (e >= N_EDGES) return;
  atomicAdd(&deg[ei[N_EDGES + e]], 1.0f);
}

__global__ __launch_bounds__(256) void deg_inv_kernel(float* __restrict__ deg) {
  int n = blockIdx.x * blockDim.x + threadIdx.x;
  if (n >= N_NODES) return;
  deg[n] = 1.0f / fmaxf(deg[n], 1.0f);
}

// ---------------------------------------------------------------------------
// Edge scatter: nbr[dst] += h[src].  One wave per edge, float4 per lane,
// 4 global f32 atomics (L2-resolved) per lane.
// ---------------------------------------------------------------------------
__global__ __launch_bounds__(256) void edge_scatter_kernel(
    const int* __restrict__ ei, const float* __restrict__ h,
    float* __restrict__ nbr) {
  long long t = (long long)blockIdx.x * blockDim.x + threadIdx.x;
  long long e = t >> 5;
  int lane = (int)(t & 31);
  if (e >= N_EDGES) return;
  int src = ei[e];
  int dst = ei[N_EDGES + e];
  const float4 v = *(const float4*)(h + (size_t)src * HIDDEN + lane * 4);
  float* p = nbr + (size_t)dst * HIDDEN + lane * 4;
  atomicAdd(p + 0, v.x);
  atomicAdd(p + 1, v.y);
  atomicAdd(p + 2, v.z);
  atomicAdd(p + 3, v.w);
}

// ---------------------------------------------------------------------------
// Fused dual GEMM via WMMA bf16:
//   tmp = (nbr * invdeg) @ Wl^T + h @ Wr^T + bl
// as one K=256 GEMM: A = [nbr*invdeg | h], B = [Wl^T ; Wr^T].
//
// Block = 256 threads = 8 waves. The full 256x128 bf16 weight matrix is
// staged once per block in LDS, pre-swizzled into WMMA B-fragment order
// (8 col-tiles x 8 k-chunks x 32 lanes x 16 bf16 = 64 KB). Each wave then
// computes a 16x128 row strip: per k-chunk it builds one A fragment from
// global (folding 1/deg into the neighbor half) and sweeps all 8 column
// tiles with ds_load'ed B fragments -> 64 v_wmma per wave.
// ---------------------------------------------------------------------------
__global__ __launch_bounds__(256) void gemm_dual_wmma_kernel(
    const float* __restrict__ nbr, const float* __restrict__ h,
    const float* __restrict__ invdeg,
    const float* __restrict__ Wl, const float* __restrict__ Wr,
    const float* __restrict__ bl, float* __restrict__ tmp) {
  __shared__ __bf16 lds[8 * 8 * 32 * 16];   // 64 KB, B fragments

  int tid = threadIdx.x;

  // ---- stage weights into LDS in fragment order -------------------------
  // flat idx bits: [14:12]=ct, [11:9]=kc, [8:4]=lane, [3:0]=i
  for (int idx = tid; idx < 8 * 8 * 32 * 16; idx += 256) {
    int i    = idx & 15;
    int lane = (idx >> 4) & 31;
    int kc   = (idx >> 9) & 7;
    int ct   = idx >> 12;
    int k    = (kc & 3) * 32 + (i & 7) + ((i >> 3) << 4) + ((lane >> 4) << 3);
    int col  = ct * 16 + (lane & 15);
    float w  = (kc < 4) ? Wl[(size_t)col * HIDDEN + k]
                        : Wr[(size_t)col * HIDDEN + k];
    lds[idx] = (__bf16)w;
  }
  __syncthreads();

  // ---- compute ----------------------------------------------------------
  int waveId = tid >> 5;
  int lane   = tid & 31;
  int sub    = lane & 15;
  int half   = lane >> 4;
  int rowTile = blockIdx.x * 8 + waveId;

  if (rowTile < ROW_TILES) {                    // wave-uniform guard
    int arow = rowTile * 16 + sub;
    float idg = invdeg[arow];

    v8f acc[8] = {};
#pragma unroll
    for (int kc = 0; kc < 8; ++kc) {
      const float* Asrc;
      float ascale;
      int kbase = (kc & 3) * 32;
      if (kc < 4) {                              // neighbor branch (lin_l)
        Asrc = nbr + (size_t)arow * HIDDEN;
        ascale = idg;                            // fold mean division
      } else {                                   // root branch (lin_r)
        Asrc = h + (size_t)arow * HIDDEN;
        ascale = 1.0f;
      }
      v16bf a;
#pragma unroll
      for (int i = 0; i < 16; ++i) {
        int k = kbase + (i & 7) + ((i >> 3) << 4) + (half << 3);
        a[i] = (__bf16)(Asrc[k] * ascale);
      }
#pragma unroll
      for (int ct = 0; ct < 8; ++ct) {
        const v16bf b =
            *(const v16bf*)&lds[((ct * 8 + kc) * 32 + lane) * 16];
        acc[ct] = __builtin_amdgcn_wmma_f32_16x16x32_bf16(
            false, a, false, b, (short)0, acc[ct], false, false);
      }
    }

#pragma unroll
    for (int ct = 0; ct < 8; ++ct) {
      int bcol = ct * 16 + sub;
      float bias = bl[bcol];
#pragma unroll
      for (int v = 0; v < 8; ++v) {
        int r = rowTile * 16 + v + (half << 3);  // C/D: M = v + 8*half
        tmp[(size_t)r * HIDDEN + bcol] = acc[ct][v] + bias;
      }
    }
  }
}

// ---------------------------------------------------------------------------
// BatchNorm statistics: per-channel sum and sum-of-squares over all nodes.
// ---------------------------------------------------------------------------
__global__ __launch_bounds__(128) void bn_stats_kernel(
    const float* __restrict__ t, float* __restrict__ bnsum,
    float* __restrict__ bnsq) {
  int c = threadIdx.x;
  float s = 0.0f, q = 0.0f;
  for (int r = blockIdx.x; r < N_NODES; r += gridDim.x) {
    float v = t[(size_t)r * HIDDEN + c];
    s += v;
    q += v * v;
  }
  atomicAdd(&bnsum[c], s);
  atomicAdd(&bnsq[c], q);
}

// ---------------------------------------------------------------------------
// Fused BN-apply + ReLU + residual:  h = act(BN(tmp)) + h
// ---------------------------------------------------------------------------
__global__ __launch_bounds__(256) void bn_apply_kernel(
    const float* __restrict__ t, float* __restrict__ h,
    const float* __restrict__ bnsum, const float* __restrict__ bnsq,
    const float* __restrict__ gamma, const float* __restrict__ beta,
    int do_relu) {
  size_t i = (size_t)blockIdx.x * blockDim.x + threadIdx.x;
  if (i >= (size_t)N_NODES * HIDDEN) return;
  int c = (int)(i & (HIDDEN - 1));
  const float invN = 1.0f / (float)N_NODES;
  float mu  = bnsum[c] * invN;
  float var = bnsq[c] * invN - mu * mu;
  float inv = rsqrtf(var + BN_EPS);
  float v = (t[i] - mu) * inv * gamma[c] + beta[c];
  if (do_relu) v = fmaxf(v, 0.0f);
  h[i] = v + h[i];
}

// ---------------------------------------------------------------------------
// Graph readout accumulate: g[b] += h[n]; gcnt[b] += 1
// ---------------------------------------------------------------------------
__global__ __launch_bounds__(128) void readout_acc_kernel(
    const int* __restrict__ batch, const float* __restrict__ h,
    float* __restrict__ g, float* __restrict__ gcnt) {
  int n = blockIdx.x;
  int c = threadIdx.x;
  int b = batch[n];
  atomicAdd(&g[(size_t)b * HIDDEN + c], h[(size_t)n * HIDDEN + c]);
  if (c == 0) atomicAdd(&gcnt[b], 1.0f);
}

// ---------------------------------------------------------------------------
// Final head: out[gi] = sigmoid(dot(g[gi]/cnt, lin_w) + lin_b)
// ---------------------------------------------------------------------------
__global__ __launch_bounds__(128) void final_kernel(
    const float* __restrict__ g, const float* __restrict__ gcnt,
    const float* __restrict__ lw, const float* __restrict__ lb,
    float* __restrict__ out) {
  __shared__ float red[128];
  int gi = blockIdx.x;
  int c = threadIdx.x;
  float cnt = fmaxf(gcnt[gi], 1.0f);
  red[c] = g[(size_t)gi * HIDDEN + c] / cnt * lw[c];
  __syncthreads();
#pragma unroll
  for (int s = 64; s > 0; s >>= 1) {
    if (c < s) red[c] += red[c + s];
    __syncthreads();
  }
  if (c == 0) out[gi] = 1.0f / (1.0f + expf(-(red[0] + lb[0])));
}

// ---------------------------------------------------------------------------
extern "C" void kernel_launch(void* const* d_in, const int* in_sizes, int n_in,
                              void* d_out, int out_size, void* d_ws,
                              size_t ws_size, hipStream_t stream) {
  const int*   x     = (const int*)d_in[0];
  const int*   ei    = (const int*)d_in[1];
  const int*   batch = (const int*)d_in[2];
  const float* emb   = (const float*)d_in[3];
  const float* Wl    = (const float*)d_in[4];
  const float* bl    = (const float*)d_in[5];
  const float* Wr    = (const float*)d_in[6];
  const float* gamma = (const float*)d_in[7];
  const float* beta  = (const float*)d_in[8];
  const float* lw    = (const float*)d_in[9];
  const float* lb    = (const float*)d_in[10];
  float* out = (float*)d_out;

  const size_t HN = (size_t)N_NODES * HIDDEN;           // 6.4M floats
  char* ws = (char*)d_ws;
  float* h      = (float*)ws;                ws += HN * sizeof(float);
  float* nbr    = (float*)ws;                ws += HN * sizeof(float);
  float* tmp    = (float*)ws;                ws += HN * sizeof(float);
  float* invdeg = (float*)ws;                ws += N_NODES * sizeof(float);
  float* bnsum  = (float*)ws;                ws += HIDDEN * sizeof(float);
  float* bnsq   = (float*)ws;                ws += HIDDEN * sizeof(float);
  float* g      = (float*)ws;                ws += (size_t)N_GRAPHS * HIDDEN * sizeof(float);
  float* gcnt   = (float*)ws;                ws += N_GRAPHS * sizeof(float);

  // Degrees (edges invariant across layers): count once, invert once.
  hipMemsetAsync(invdeg, 0, N_NODES * sizeof(float), stream);
  deg_count_kernel<<<(N_EDGES + 255) / 256, 256, 0, stream>>>(ei, invdeg);
  deg_inv_kernel<<<(N_NODES + 255) / 256, 256, 0, stream>>>(invdeg);

  // AtomEncoder
  atom_encode_kernel<<<N_NODES, HIDDEN, 0, stream>>>(x, emb, h);

  const long long scatter_threads = (long long)N_EDGES * 32;
  const int scatter_blocks = (int)((scatter_threads + 255) / 256);
  const int gemm_blocks = (ROW_TILES + 7) / 8;          // 391

  for (int l = 0; l < N_LAYERS; ++l) {
    hipMemsetAsync(nbr, 0, HN * sizeof(float), stream);
    edge_scatter_kernel<<<scatter_blocks, 256, 0, stream>>>(ei, h, nbr);

    gemm_dual_wmma_kernel<<<gemm_blocks, 256, 0, stream>>>(
        nbr, h, invdeg,
        Wl + (size_t)l * HIDDEN * HIDDEN, Wr + (size_t)l * HIDDEN * HIDDEN,
        bl + (size_t)l * HIDDEN, tmp);

    hipMemsetAsync(bnsum, 0, 2 * HIDDEN * sizeof(float), stream);  // bnsum+bnsq
    bn_stats_kernel<<<256, HIDDEN, 0, stream>>>(tmp, bnsum, bnsq);

    bn_apply_kernel<<<(int)((HN + 255) / 256), 256, 0, stream>>>(
        tmp, h, bnsum, bnsq, gamma + (size_t)l * HIDDEN,
        beta + (size_t)l * HIDDEN, (l < N_LAYERS - 1) ? 1 : 0);
  }

  // Readout + head
  hipMemsetAsync(g, 0, ((size_t)N_GRAPHS * HIDDEN + N_GRAPHS) * sizeof(float),
                 stream);
  readout_acc_kernel<<<N_NODES, HIDDEN, 0, stream>>>(batch, h, g, gcnt);
  final_kernel<<<N_GRAPHS, HIDDEN, 0, stream>>>(g, gcnt, lw, lb, out);
}
